// HANRegression_13597866459799
// MI455X (gfx1250) — compile-verified, matrix operand order checked by reading, not run on hardware
//
#include <hip/hip_runtime.h>
#include <hip/hip_bf16.h>

typedef __attribute__((ext_vector_type(16))) _Float16 v16h;
typedef __attribute__((ext_vector_type(8)))  float    v8f;

#define N_JOB   100000
#define N_SKILL 20000
#define E_SJ    500000
#define E_JJ    500000

static inline int cdiv(long a, long b) { return (int)((a + b - 1) / b); }

__device__ __forceinline__ float atomic_max_float(float* addr, float val) {
  if (val >= 0.f)
    return __int_as_float(atomicMax((int*)addr, __float_as_int(val)));
  else
    return __uint_as_float(atomicMin((unsigned int*)addr, __float_as_uint(val)));
}

// ---------------------------------------------------------------------------
// C[N x 256] = op( A[N x 256] @ W[256 x 256] + bias )
// Block: 256 threads = 8 waves, covers 128 rows x 64 cols. K looped 256/32.
// MODE 0: store bias-added; MODE 1: store relu; MODE 2: no store, reduce
//         sum(q[f] * tanh(c[n,f]+bias[f])) into *scoreOut via atomicAdd.
// Fragment layouts per CDNA5 ISA 7.12.2 (16-bit A 16x32; B mirrors A over
// columns; C/D 8 VGPRs, M = r + 8*(lane>=16), N = lane&15).
// ---------------------------------------------------------------------------
template<int MODE>
__global__ __launch_bounds__(256) void gemm256_kernel(
    const float* __restrict__ A, const float* __restrict__ W,
    const float* __restrict__ bias, float* __restrict__ C,
    const float* __restrict__ qv, float* __restrict__ scoreOut, int N)
{
  __shared__ _Float16 sW[64 * 260];   // column-major [f][k], pad 4 halfs
  const int tid  = threadIdx.x;
  const int lane = tid & 31;
  const int wave = tid >> 5;
  const int nb   = blockIdx.y * 64;

  // stage 256x64 slice of W into LDS as f16 (coalesced global reads)
  for (int i = tid; i < 64 * 256; i += 256) {
    int f = i & 63, k = i >> 6;
    sW[f * 260 + k] = (_Float16)W[(size_t)k * 256 + nb + f];
  }
  __syncthreads();

  const int m0 = blockIdx.x * 128 + wave * 16;
  if (m0 >= N) return;                  // wave-uniform (N % 16 == 0)

  const int col   = lane & 15;
  const int khalf = (lane >> 4) << 3;   // 0 for lanes 0-15, 8 for 16-31

  v8f acc[4] = {};
  const float* ap = A + (size_t)(m0 + col) * 256 + khalf;

  for (int k0 = 0; k0 < 256; k0 += 32) {
    union { v16h v; _Float16 h[16]; } afrag;
#pragma unroll
    for (int e = 0; e < 16; ++e) {
      int kk = ((e >> 3) << 4) + (e & 7);           // {0..7,16..23} rel.
      afrag.h[e] = (_Float16)ap[k0 + kk];
    }
#pragma unroll
    for (int t = 0; t < 4; ++t) {
      union { v16h v; _Float16 h[16]; } bfrag;
      const _Float16* bp = &sW[(t * 16 + col) * 260 + k0 + khalf];
#pragma unroll
      for (int e = 0; e < 16; ++e) {
        int kk = ((e >> 3) << 4) + (e & 7);
        bfrag.h[e] = bp[kk];
      }
      acc[t] = __builtin_amdgcn_wmma_f32_16x16x32_f16(
          false, afrag.v, false, bfrag.v, (short)0, acc[t], false, false);
    }
  }

  if (MODE == 2) {
    float s = 0.f;
#pragma unroll
    for (int t = 0; t < 4; ++t) {
      int cg = nb + t * 16 + col;
      float b = bias[cg], qf = qv[cg];
#pragma unroll
      for (int r = 0; r < 8; ++r) s += qf * tanhf(acc[t][r] + b);
    }
#pragma unroll
    for (int m = 16; m >= 1; m >>= 1) s += __shfl_xor(s, m, 32);
    if (lane == 0) atomicAdd(scoreOut, s);
  } else {
#pragma unroll
    for (int t = 0; t < 4; ++t) {
      int cg = nb + t * 16 + col;
      float b = bias[cg];
#pragma unroll
      for (int r = 0; r < 8; ++r) {
        float v = acc[t][r] + b;
        if (MODE == 1) v = fmaxf(v, 0.f);
        C[(size_t)(m0 + r + khalf) * 256 + cg] = v;
      }
    }
  }
}

// ---------------------------------------------------------------------------
// GAT building blocks
// ---------------------------------------------------------------------------
__global__ void node_logits_kernel(const float* __restrict__ X,
                                   const float* __restrict__ avec,
                                   float* __restrict__ out, int N)
{
  int t = blockIdx.x * blockDim.x + threadIdx.x;     // t = n*4 + h
  if (t >= N * 4) return;
  int n = t >> 2, h = t & 3;
  const float* x = X + (size_t)n * 256 + h * 64;
  const float* a = avec + h * 64;
  float s = 0.f;
#pragma unroll 16
  for (int d = 0; d < 64; ++d) s += x[d] * a[d];
  out[t] = s;
}

__global__ void fill_u32_kernel(unsigned int* __restrict__ p, unsigned int v, long n) {
  long i = (long)blockIdx.x * blockDim.x + threadIdx.x;
  if (i < n) p[i] = v;
}

__global__ void edge_logits_kernel(const int* __restrict__ src, const int* __restrict__ dst,
                                   const float* __restrict__ al_s, const float* __restrict__ al_d,
                                   float* __restrict__ ebuf, float* __restrict__ amax, int E)
{
  int t = blockIdx.x * blockDim.x + threadIdx.x;     // t = e*4 + h
  if (t >= E * 4) return;
  int e = t >> 2, h = t & 3;
  float v = al_s[src[e] * 4 + h] + al_d[dst[e] * 4 + h];
  v = (v > 0.f) ? v : 0.2f * v;                      // leaky_relu(0.2)
  ebuf[t] = v;
  atomic_max_float(&amax[dst[e] * 4 + h], v);
}

__global__ void edge_exp_kernel(const int* __restrict__ dst, const float* __restrict__ amax,
                                float* __restrict__ ebuf, float* __restrict__ denom, int E)
{
  int t = blockIdx.x * blockDim.x + threadIdx.x;
  if (t >= E * 4) return;
  int e = t >> 2, h = t & 3;
  float v = __expf(ebuf[t] - amax[dst[e] * 4 + h]);
  ebuf[t] = v;
  atomicAdd(&denom[dst[e] * 4 + h], v);
}

__global__ __launch_bounds__(256) void edge_scatter_kernel(
    const int* __restrict__ src, const int* __restrict__ dst,
    const float* __restrict__ ebuf, const float* __restrict__ denom,
    const float* __restrict__ Xs, float* __restrict__ out, int E)
{
  int e = blockIdx.x;          // one edge per block
  int c = threadIdx.x;         // channel 0..255
  int h = c >> 6;
  int s = src[e], d = dst[e];
  float alpha = ebuf[e * 4 + h] / (denom[d * 4 + h] + 1e-16f);
  atomicAdd(&out[(size_t)d * 256 + c], Xs[(size_t)s * 256 + c] * alpha);
}

__global__ void relu_kernel(float* __restrict__ p, long n) {
  long i = (long)blockIdx.x * blockDim.x + threadIdx.x;
  if (i < n) p[i] = fmaxf(p[i], 0.f);
}

// job = attn0*out_sj + attn1*out_jj  (softmax over 2 per-type scores, in place)
__global__ void combine_kernel(const float* __restrict__ scores, float* __restrict__ a,
                               const float* __restrict__ b, long n, float invN)
{
  long i = (long)blockIdx.x * blockDim.x + threadIdx.x;
  if (i >= n) return;
  float s0 = scores[0] * invN, s1 = scores[1] * invN;
  float m  = fmaxf(s0, s1);
  float e0 = __expf(s0 - m), e1 = __expf(s1 - m);
  float inv = 1.f / (e0 + e1);
  a[i] = (e0 * inv) * a[i] + (e1 * inv) * b[i];
}

// out[n] = hidden[n,:] . W2 + b2   (one wave per node, shfl reduce)
__global__ __launch_bounds__(256) void head_kernel(
    const float* __restrict__ Hm, const float* __restrict__ W2,
    const float* __restrict__ b2, float* __restrict__ out, int N)
{
  int lane = threadIdx.x & 31;
  int node = blockIdx.x * 8 + (threadIdx.x >> 5);
  if (node >= N) return;
  const float* hp = Hm + (size_t)node * 256 + lane * 8;
  const float* wp = W2 + lane * 8;
  float s = 0.f;
#pragma unroll
  for (int j = 0; j < 8; ++j) s += hp[j] * wp[j];
#pragma unroll
  for (int m = 16; m >= 1; m >>= 1) s += __shfl_xor(s, m, 32);
  if (lane == 0) out[node] = s + b2[0];
}

// ---------------------------------------------------------------------------
static void run_gat(const float* Xsrc, int Ns, const float* Xdst, int Nd,
                    const float* a_src, const float* a_dst,
                    const int* esrc, const int* edst, int E,
                    float* al_s, float* al_d, float* amax, float* denom,
                    float* ebuf, float* out, hipStream_t stream)
{
  node_logits_kernel<<<cdiv((long)Ns * 4, 256), 256, 0, stream>>>(Xsrc, a_src, al_s, Ns);
  node_logits_kernel<<<cdiv((long)Nd * 4, 256), 256, 0, stream>>>(Xdst, a_dst, al_d, Nd);
  fill_u32_kernel<<<cdiv((long)Nd * 4, 256), 256, 0, stream>>>((unsigned*)amax, 0xFF800000u, (long)Nd * 4);
  fill_u32_kernel<<<cdiv((long)Nd * 4, 256), 256, 0, stream>>>((unsigned*)denom, 0u, (long)Nd * 4);
  fill_u32_kernel<<<cdiv((long)Nd * 256, 256), 256, 0, stream>>>((unsigned*)out, 0u, (long)Nd * 256);
  edge_logits_kernel<<<cdiv((long)E * 4, 256), 256, 0, stream>>>(esrc, edst, al_s, al_d, ebuf, amax, E);
  edge_exp_kernel<<<cdiv((long)E * 4, 256), 256, 0, stream>>>(edst, amax, ebuf, denom, E);
  edge_scatter_kernel<<<E, 256, 0, stream>>>(esrc, edst, ebuf, denom, Xsrc, out, E);
  relu_kernel<<<cdiv((long)Nd * 256, 256), 256, 0, stream>>>(out, (long)Nd * 256);
}

extern "C" void kernel_launch(void* const* d_in, const int* in_sizes, int n_in,
                              void* d_out, int out_size, void* d_ws, size_t ws_size,
                              hipStream_t stream)
{
  (void)in_sizes; (void)n_in; (void)out_size; (void)ws_size;
  const float* x_job    = (const float*)d_in[0];
  const float* x_skill  = (const float*)d_in[1];
  const float* W_job    = (const float*)d_in[2];
  const float* b_job    = (const float*)d_in[3];
  const float* W_skill  = (const float*)d_in[4];
  const float* b_skill  = (const float*)d_in[5];
  const float* a_src_sj = (const float*)d_in[6];
  const float* a_dst_sj = (const float*)d_in[7];
  const float* a_src_jj = (const float*)d_in[8];
  const float* a_dst_jj = (const float*)d_in[9];
  // d_in[10], d_in[11]: a_src_js / a_dst_js -- feed only the dead skill branch
  const float* q  = (const float*)d_in[12];
  const float* Wk = (const float*)d_in[13];
  const float* bk = (const float*)d_in[14];
  const float* W1 = (const float*)d_in[15];
  const float* b1 = (const float*)d_in[16];
  const float* W2 = (const float*)d_in[17];
  const float* b2 = (const float*)d_in[18];
  const int* ei_sj_src = (const int*)d_in[19];
  const int* ei_sj_dst = (const int*)d_in[20];
  const int* ei_jj_src = (const int*)d_in[21];
  const int* ei_jj_dst = (const int*)d_in[22];
  // d_in[23], d_in[24]: js edges -- dead branch, skipped

  float* ws = (float*)d_ws;
  float* h_job   = ws; ws += (size_t)N_JOB   * 256;
  float* h_skill = ws; ws += (size_t)N_SKILL * 256;
  float* out_sj  = ws; ws += (size_t)N_JOB   * 256;
  float* out_jj  = ws; ws += (size_t)N_JOB   * 256;
  float* ebuf    = ws; ws += (size_t)E_SJ    * 4;
  float* al_s    = ws; ws += (size_t)N_JOB   * 4;
  float* al_d    = ws; ws += (size_t)N_JOB   * 4;
  float* amax    = ws; ws += (size_t)N_JOB   * 4;
  float* denom   = ws; ws += (size_t)N_JOB   * 4;
  float* scores  = ws;

  dim3 blk(256);

  // 1. feature projections (f16 WMMA, f32 accumulate)
  gemm256_kernel<0><<<dim3(cdiv(N_JOB, 128), 4), blk, 0, stream>>>(
      x_job, W_job, b_job, h_job, nullptr, nullptr, N_JOB);
  gemm256_kernel<0><<<dim3(cdiv(N_SKILL, 128), 4), blk, 0, stream>>>(
      x_skill, W_skill, b_skill, h_skill, nullptr, nullptr, N_SKILL);

  // 2. GAT skill->job and job->job (job->skill result is never consumed)
  run_gat(h_skill, N_SKILL, h_job, N_JOB, a_src_sj, a_dst_sj,
          ei_sj_src, ei_sj_dst, E_SJ, al_s, al_d, amax, denom, ebuf, out_sj, stream);
  run_gat(h_job, N_JOB, h_job, N_JOB, a_src_jj, a_dst_jj,
          ei_jj_src, ei_jj_dst, E_JJ, al_s, al_d, amax, denom, ebuf, out_jj, stream);

  // 3. semantic attention scores: fused GEMM + tanh + q-dot reduction
  fill_u32_kernel<<<1, 2, 0, stream>>>((unsigned*)scores, 0u, 2);
  gemm256_kernel<2><<<dim3(cdiv(N_JOB, 128), 4), blk, 0, stream>>>(
      out_sj, Wk, bk, nullptr, q, scores + 0, N_JOB);
  gemm256_kernel<2><<<dim3(cdiv(N_JOB, 128), 4), blk, 0, stream>>>(
      out_jj, Wk, bk, nullptr, q, scores + 1, N_JOB);

  // 4. softmax(2)-weighted combine, in place into out_sj
  combine_kernel<<<cdiv((long)N_JOB * 256, 256), blk, 0, stream>>>(
      scores, out_sj, out_jj, (long)N_JOB * 256, 1.f / (float)N_JOB);

  // 5. MLP hidden = relu(job @ W1 + b1)  (reuse out_jj as hidden buffer)
  gemm256_kernel<1><<<dim3(cdiv(N_JOB, 128), 4), blk, 0, stream>>>(
      out_sj, W1, b1, out_jj, nullptr, nullptr, N_JOB);

  // 6. regression head: out = hidden @ W2 + b2
  head_kernel<<<cdiv(N_JOB, 8), blk, 0, stream>>>(out_jj, W2, b2, (float*)d_out, N_JOB);
}
